// GRUWorldModel_81647328297666
// MI455X (gfx1250) — matrix operand order repaired and checked by reading, not compiled, not run on hardware
//
#include <hip/hip_runtime.h>

typedef __attribute__((ext_vector_type(16))) _Float16 v16h;
typedef __attribute__((ext_vector_type(8)))  _Float16 v8h;
typedef __attribute__((ext_vector_type(8)))  float    v8f;

// ---------------- f16 weight workspace layout (units: halves) ----------------
#define OF_FUSE 0         // [128][96]  (K padded 80 -> 96 with zeros)
#define OF_IH   12288     // [384][128]
#define OF_HH   61440     // [384][128]
#define OF_NS1  110592    // [128][128]
#define OF_NS2  126976    // [64][128]
#define OF_RW1  135168    // [64][128]
#define OF_DN1  143360    // [64][128]
#define WS_HALVES 151552  // ~303 KB

#define LDS_PITCH 136     // halves per row of per-wave activation buffer (pad vs bank conflicts)

// ---------------- WMMA helpers ----------------
__device__ __forceinline__ v8f wmma_f16(v16h a, v16h b, v8f c) {
  // D = A(16x32 f16) x B(32x16 f16) + C(16x16 f32)
  return __builtin_amdgcn_wmma_f32_16x16x32_f16(false, a, false, b, (short)0, c, false, false);
}

__device__ __forceinline__ v8f splat8(float x) {
  v8f v;
#pragma unroll
  for (int r = 0; r < 8; ++r) v[r] = x;
  return v;
}

__device__ __forceinline__ v16h pack_a(v8f lo, v8f hv) {
  v8h l = __builtin_convertvector(lo, v8h);
  v8h h = __builtin_convertvector(hv, v8h);
  v16h a;
#pragma unroll
  for (int j = 0; j < 8; ++j) { a[j] = l[j]; a[j + 8] = h[j]; }
  return a;
}

// A fragment (16x32) from row-major f32 matrix. Lane (c,hi) holds row c,
// K = k0+8*hi+j (j<8) and k0+16+8*hi+(j-8) (j>=8)   [CDNA5 16-bit A layout]
__device__ __forceinline__ v16h load_a_g(const float* __restrict__ row, int k0, int hi) {
  v8f lo = *(const v8f*)(row + k0 + 8 * hi);
  v8f hv = *(const v8f*)(row + k0 + 8 * hi + 16);
  return pack_a(lo, hv);
}

// B fragment (32x16): column n = weight row (n0+c); lane (c,hi) holds
// halves K = k0+8*hi+j (j<8) and k0+16+8*hi+(j-8) (j>=8) of the f16 weight row.
__device__ __forceinline__ v16h load_b_g(const _Float16* __restrict__ W, int K,
                                         int n0, int c, int k0, int hi) {
  const _Float16* p = W + (size_t)(n0 + c) * K + k0 + 8 * hi;
  v8h l = *(const v8h*)p;
  v8h h = *(const v8h*)(p + 16);
  v16h b;
#pragma unroll
  for (int j = 0; j < 8; ++j) { b[j] = l[j]; b[j + 8] = h[j]; }
  return b;
}

// A fragment from per-wave LDS activation buffer (row-major f16, pitch LDS_PITCH)
__device__ __forceinline__ v16h load_a_lds(const _Float16* buf, int c, int k0, int hi) {
  const _Float16* p = buf + c * LDS_PITCH + k0 + 8 * hi;
  v8h l = *(const v8h*)p;
  v8h h = *(const v8h*)(p + 16);
  v16h a;
#pragma unroll
  for (int j = 0; j < 8; ++j) { a[j] = l[j]; a[j + 8] = h[j]; }
  return a;
}

__device__ __forceinline__ float sigmoid_f(float x) { return 1.f / (1.f + __expf(-x)); }
__device__ __forceinline__ float tanh_f(float x)    { return 1.f - 2.f / (__expf(2.f * x) + 1.f); }

// XOR butterfly within the 16-lane column group (wave32 ds_swizzle)
template <int M>
__device__ __forceinline__ float xshuf(float v) {
  return __int_as_float(__builtin_amdgcn_ds_swizzle(__float_as_int(v), (M << 10) | 0x1f));
}
__device__ __forceinline__ float red16(float v) {
  v += xshuf<1>(v); v += xshuf<2>(v); v += xshuf<4>(v); v += xshuf<8>(v);
  return v;
}

// ---------------- weight f32 -> f16 prep ----------------
__global__ void gru_prep(const float* __restrict__ fw,  // [128][80]
                         const float* __restrict__ wi,  // [384][128]
                         const float* __restrict__ wh,  // [384][128]
                         const float* __restrict__ n1,  // [128][128]
                         const float* __restrict__ n2,  // [64][128]
                         const float* __restrict__ r1,  // [64][128]
                         const float* __restrict__ d1,  // [64][128]
                         _Float16* __restrict__ ws) {
  const int i0 = blockIdx.x * blockDim.x + threadIdx.x;
  const int np = gridDim.x * blockDim.x;
  for (int i = i0; i < 128 * 96; i += np) {
    int r = i / 96, c = i % 96;
    ws[OF_FUSE + i] = (_Float16)(c < 80 ? fw[r * 80 + c] : 0.f);
  }
  for (int i = i0; i < 384 * 128; i += np) ws[OF_IH + i] = (_Float16)wi[i];
  for (int i = i0; i < 384 * 128; i += np) ws[OF_HH + i] = (_Float16)wh[i];
  for (int i = i0; i < 128 * 128; i += np) ws[OF_NS1 + i] = (_Float16)n1[i];
  for (int i = i0; i <  64 * 128; i += np) ws[OF_NS2 + i] = (_Float16)n2[i];
  for (int i = i0; i <  64 * 128; i += np) ws[OF_RW1 + i] = (_Float16)r1[i];
  for (int i = i0; i <  64 * 128; i += np) ws[OF_DN1 + i] = (_Float16)d1[i];
}

// ---------------- main fused GRU world-model kernel ----------------
__global__ __launch_bounds__(256) void gru_main(
    const float* __restrict__ z_t,     // [B][64]
    const int*   __restrict__ action,  // [B]
    const float* __restrict__ hidden,  // [B][128]
    const float* __restrict__ embed,   // [9][16]
    const float* __restrict__ fuse_b,  // [128]
    const float* __restrict__ b_ih,    // [384]
    const float* __restrict__ b_hh,    // [384]
    const float* __restrict__ ns_b1,   // [128]
    const float* __restrict__ ns_b2,   // [64]
    const float* __restrict__ rw_b1,   // [64]
    const float* __restrict__ rw_W2,   // [64]
    const float* __restrict__ rw_b2,   // [1]
    const float* __restrict__ dn_b1,   // [64]
    const float* __restrict__ dn_W2,   // [64]
    const float* __restrict__ dn_b2,   // [1]
    const _Float16* __restrict__ ws,
    float* __restrict__ out, int B) {
  __shared__ _Float16 smem[8 * 16 * LDS_PITCH];

  const int w    = threadIdx.x >> 5;
  const int lane = threadIdx.x & 31;
  const int c    = lane & 15;   // column within 16x16 tile
  const int hi   = lane >> 4;   // half-wave selector
  _Float16* buf = smem + w * (16 * LDS_PITCH);

  const _Float16* WF = ws + OF_FUSE;
  const _Float16* WI = ws + OF_IH;
  const _Float16* WH = ws + OF_HH;
  const _Float16* N1 = ws + OF_NS1;
  const _Float16* N2 = ws + OF_NS2;
  const _Float16* R1 = ws + OF_RW1;
  const _Float16* D1 = ws + OF_DN1;

  float* out_z = out;                          // [B][64]
  float* out_r = out + (size_t)B * 64;         // [B]
  float* out_d = out + (size_t)B * 65;         // [B]
  float* out_h = out + (size_t)B * 66;         // [B][128]

  // per-lane head-2 weight columns (reused for every tile)
  float rw2c[4], dn2c[4];
#pragma unroll
  for (int t = 0; t < 4; ++t) { rw2c[t] = rw_W2[16 * t + c]; dn2c[t] = dn_W2[16 * t + c]; }
  const float rb2 = rw_b2[0], db2 = dn_b2[0];

  const int ntiles = B >> 4;
  const int tstep  = gridDim.x * 8;
  for (int tile = blockIdx.x * 8 + w; tile < ntiles; tile += tstep) {
    const int m0    = tile << 4;
    const int myrow = m0 + c;

    // prefetch the next grid-stride tile's streaming rows (global_prefetch_b8)
    {
      const int nt = tile + tstep;
      if (nt < ntiles) {
        const int nrow = (nt << 4) + c;
        __builtin_prefetch(z_t + (size_t)nrow * 64 + 32 * hi, 0, 3);
        __builtin_prefetch(hidden + (size_t)nrow * 128 + 64 * hi, 0, 3);
      }
    }

    // ---- fused = [z_t | embed[action] | 0-pad] : A fragments, K = 96
    const float* zrow = z_t + (size_t)myrow * 64;
    v16h aF0 = load_a_g(zrow, 0, hi);
    v16h aF1 = load_a_g(zrow, 32, hi);
    v16h aF2;
    {
      int a = action[myrow];
      v8f ev = *(const v8f*)(embed + a * 16 + 8 * hi);
      v8h el = __builtin_convertvector(ev, v8h);
#pragma unroll
      for (int j = 0; j < 8; ++j) { aF2[j] = el[j]; aF2[j + 8] = (_Float16)0.f; }
    }

    // ---- x = relu(fused @ fuse_W^T + fuse_b)  -> LDS -> A fragments
    // unroll 2: two independent accumulator chains hide WMMA->WMMA hazard slots
#pragma unroll 2
    for (int t = 0; t < 8; ++t) {
      v8f acc = splat8(fuse_b[16 * t + c]);
      acc = wmma_f16(aF0, load_b_g(WF, 96, 16 * t, c, 0, hi), acc);
      acc = wmma_f16(aF1, load_b_g(WF, 96, 16 * t, c, 32, hi), acc);
      acc = wmma_f16(aF2, load_b_g(WF, 96, 16 * t, c, 64, hi), acc);
#pragma unroll
      for (int r = 0; r < 8; ++r)
        buf[(r + 8 * hi) * LDS_PITCH + 16 * t + c] = (_Float16)fmaxf(acc[r], 0.f);
    }
    v16h aX[4];
#pragma unroll
    for (int ks = 0; ks < 4; ++ks) aX[ks] = load_a_lds(buf, c, 32 * ks, hi);

    // ---- h0 A fragments straight from global (row-major)
    const float* hrow = hidden + (size_t)myrow * 128;
    v16h aH[4];
#pragma unroll
    for (int ks = 0; ks < 4; ++ks) aH[ks] = load_a_g(hrow, 32 * ks, hi);

    // ---- GRU gates, tile-by-tile over the 128 hidden cols
    // (4 independent accumulator chains already interleave per iteration)
#pragma unroll 1
    for (int t = 0; t < 8; ++t) {
      const int n = 16 * t + c;
      v8f gr  = splat8(b_ih[n] + b_hh[n]);
      v8f gz  = splat8(b_ih[128 + n] + b_hh[128 + n]);
      v8f gi2 = splat8(b_ih[256 + n]);
      v8f gh2 = splat8(b_hh[256 + n]);
#pragma unroll
      for (int ks = 0; ks < 4; ++ks) {
        gr  = wmma_f16(aX[ks], load_b_g(WI, 128,       16 * t, c, 32 * ks, hi), gr);
        gr  = wmma_f16(aH[ks], load_b_g(WH, 128,       16 * t, c, 32 * ks, hi), gr);
        gz  = wmma_f16(aX[ks], load_b_g(WI, 128, 128 + 16 * t, c, 32 * ks, hi), gz);
        gz  = wmma_f16(aH[ks], load_b_g(WH, 128, 128 + 16 * t, c, 32 * ks, hi), gz);
        gi2 = wmma_f16(aX[ks], load_b_g(WI, 128, 256 + 16 * t, c, 32 * ks, hi), gi2);
        gh2 = wmma_f16(aH[ks], load_b_g(WH, 128, 256 + 16 * t, c, 32 * ks, hi), gh2);
      }
#pragma unroll
      for (int r = 0; r < 8; ++r) {
        float rg  = sigmoid_f(gr[r]);
        float zg  = sigmoid_f(gz[r]);
        float ng  = tanh_f(gi2[r] + rg * gh2[r]);
        float h0v = hidden[(size_t)(m0 + r + 8 * hi) * 128 + n];
        float hn  = (1.f - zg) * ng + zg * h0v;
        out_h[(size_t)(m0 + r + 8 * hi) * 128 + n] = hn;
        buf[(r + 8 * hi) * LDS_PITCH + n] = (_Float16)hn;  // overwrite x buffer
      }
    }
    v16h aHN[4];
#pragma unroll
    for (int ks = 0; ks < 4; ++ks) aHN[ks] = load_a_lds(buf, c, 32 * ks, hi);

    // ---- next-state head: hid1 = relu(h @ ns_W1^T + b1) -> LDS -> A frags
#pragma unroll 2
    for (int t = 0; t < 8; ++t) {
      v8f acc = splat8(ns_b1[16 * t + c]);
#pragma unroll
      for (int ks = 0; ks < 4; ++ks)
        acc = wmma_f16(aHN[ks], load_b_g(N1, 128, 16 * t, c, 32 * ks, hi), acc);
#pragma unroll
      for (int r = 0; r < 8; ++r)
        buf[(r + 8 * hi) * LDS_PITCH + 16 * t + c] = (_Float16)fmaxf(acc[r], 0.f);
    }
    v16h aG[4];
#pragma unroll
    for (int ks = 0; ks < 4; ++ks) aG[ks] = load_a_lds(buf, c, 32 * ks, hi);

    // ---- z_next = hid1 @ ns_W2^T + b2
#pragma unroll 2
    for (int t = 0; t < 4; ++t) {
      v8f acc = splat8(ns_b2[16 * t + c]);
#pragma unroll
      for (int ks = 0; ks < 4; ++ks)
        acc = wmma_f16(aG[ks], load_b_g(N2, 128, 16 * t, c, 32 * ks, hi), acc);
#pragma unroll
      for (int r = 0; r < 8; ++r)
        out_z[(size_t)(m0 + r + 8 * hi) * 64 + 16 * t + c] = acc[r];
    }

    // ---- reward & done heads: hidden layer via WMMA, final 64->1 dot via
    //      per-lane FMAs + 16-lane XOR reduction (2 chains interleave)
    float pr[8], pd[8];
#pragma unroll
    for (int r = 0; r < 8; ++r) { pr[r] = 0.f; pd[r] = 0.f; }
#pragma unroll 1
    for (int t = 0; t < 4; ++t) {
      v8f ar = splat8(rw_b1[16 * t + c]);
      v8f ad = splat8(dn_b1[16 * t + c]);
#pragma unroll
      for (int ks = 0; ks < 4; ++ks) {
        ar = wmma_f16(aHN[ks], load_b_g(R1, 128, 16 * t, c, 32 * ks, hi), ar);
        ad = wmma_f16(aHN[ks], load_b_g(D1, 128, 16 * t, c, 32 * ks, hi), ad);
      }
#pragma unroll
      for (int r = 0; r < 8; ++r) {
        pr[r] += fmaxf(ar[r], 0.f) * rw2c[t];
        pd[r] += fmaxf(ad[r], 0.f) * dn2c[t];
      }
    }
#pragma unroll
    for (int r = 0; r < 8; ++r) { pr[r] = red16(pr[r]); pd[r] = red16(pd[r]); }
    if (c == 0) {
#pragma unroll
      for (int r = 0; r < 8; ++r) {
        out_r[m0 + 8 * hi + r] = pr[r] + rb2;
        out_d[m0 + 8 * hi + r] = pd[r] + db2;
      }
    }
  }
}

extern "C" void kernel_launch(void* const* d_in, const int* in_sizes, int n_in,
                              void* d_out, int out_size, void* d_ws, size_t ws_size,
                              hipStream_t stream) {
  const float* z_t  = (const float*)d_in[0];
  const int*   act  = (const int*)d_in[1];
  const float* hid  = (const float*)d_in[2];
  const float* emb  = (const float*)d_in[3];
  const float* fW   = (const float*)d_in[4];
  const float* fb   = (const float*)d_in[5];
  const float* Wih  = (const float*)d_in[6];
  const float* Whh  = (const float*)d_in[7];
  const float* bih  = (const float*)d_in[8];
  const float* bhh  = (const float*)d_in[9];
  const float* nW1  = (const float*)d_in[10];
  const float* nb1  = (const float*)d_in[11];
  const float* nW2  = (const float*)d_in[12];
  const float* nb2  = (const float*)d_in[13];
  const float* rW1  = (const float*)d_in[14];
  const float* rb1  = (const float*)d_in[15];
  const float* rW2  = (const float*)d_in[16];
  const float* rb2  = (const float*)d_in[17];
  const float* dW1  = (const float*)d_in[18];
  const float* db1  = (const float*)d_in[19];
  const float* dW2  = (const float*)d_in[20];
  const float* db2  = (const float*)d_in[21];

  const int B = in_sizes[1];           // action is [B]
  _Float16* ws = (_Float16*)d_ws;

  gru_prep<<<240, 256, 0, stream>>>(fW, Wih, Whh, nW1, nW2, rW1, dW1, ws);

  const int ntiles = B >> 4;           // 16 batch rows per wave
  int blocks = (ntiles + 7) >> 3;      // 8 waves per block
  if (blocks > 4096) blocks = 4096;    // grid-stride covers the rest
  gru_main<<<blocks, 256, 0, stream>>>(z_t, act, hid, emb, fb, bih, bhh, nb1, nb2,
                                       rb1, rW2, rb2, db1, dW2, db2,
                                       ws, (float*)d_out, B);
}
